// Cluster_layer_36721970381457
// MI455X (gfx1250) — matrix operand order repaired and checked by reading, not compiled, not run on hardware
//
#include <hip/hip_runtime.h>

typedef __attribute__((ext_vector_type(16))) __bf16 v16bf;
typedef __attribute__((ext_vector_type(8)))  __bf16 v8bf;
typedef __attribute__((ext_vector_type(8)))  float  v8f;
typedef __attribute__((ext_vector_type(2)))  float  v2f;

#define N_PTS   196
#define DIM     512
#define NPAD    224           // n padded to 7*32 (K of einsum2), >= 13*16 = 208
#define DPITCH  528           // u row pitch: 1056B, rotates banks by 8 dwords/row
#define OBPITCH 520           // ob row pitch: 1040B, rotates banks by 4 dwords/row
#define N_ITERS 10
#define BETA_F  (-10.0f)
#define EPS_F   (1e-10f)

#define U_BYTES   (NPAD * DPITCH * 2)          // 236544: bf16 u[224][528]
#define O_OFF     (U_BYTES)                    // f32 o[16][512]
#define O_BYTES   (16 * DIM * 4)
#define OB_OFF    (O_OFF + O_BYTES)            // bf16 ob[16][520] = bf16(-2*o)
#define OB_BYTES  (16 * OBPITCH * 2)
#define C_OFF     (OB_OFF + OB_BYTES)          // f32 c[16][224]
#define C_BYTES   (16 * NPAD * 4)
#define NY_OFF    (C_OFF + C_BYTES)            // f32 ny[208]
#define NY_BYTES  (208 * 4)
#define NX_OFF    (NY_OFF + NY_BYTES)          // f32 nx[2] (padded)
#define CS_OFF    (NX_OFF + 32)                // f32 csum[2] (padded)
#define LDS_BYTES (CS_OFF + 32)                // 301184 B < 320 KB

__global__ void __launch_bounds__(256)
cluster_kernel(const float* __restrict__ u_g, float* __restrict__ out, int batches)
{
    extern __shared__ __align__(128) char smem[];
    __bf16* u_lds  = (__bf16*)smem;
    float*  o_lds  = (float*)(smem + O_OFF);
    __bf16* ob_lds = (__bf16*)(smem + OB_OFF);
    float*  c_lds  = (float*)(smem + C_OFF);
    float*  ny_lds = (float*)(smem + NY_OFF);
    float*  nx_lds = (float*)(smem + NX_OFF);
    float*  cs_lds = (float*)(smem + CS_OFF);

    const int tid  = threadIdx.x;
    const int lane = tid & 31;
    const int wave = tid >> 5;        // 0..7
    const int half = lane >> 4;       // 0: lanes 0-15, 1: lanes 16-31
    const int lrow = lane & 15;
    const int b    = blockIdx.x;
    const float* ug = u_g + (size_t)b * (N_PTS * DIM);

    // ---------------- setup: u -> LDS (bf16), zero pads, init o ----------------
    for (int idx = tid; idx < N_PTS * DIM; idx += 256) {
        int n = idx >> 9, d = idx & (DIM - 1);
        u_lds[n * DPITCH + d] = (__bf16)ug[idx];
    }
    for (int idx = tid; idx < (NPAD - N_PTS) * DIM; idx += 256) {
        int n = N_PTS + (idx >> 9), d = idx & (DIM - 1);
        u_lds[n * DPITCH + d] = (__bf16)0.0f;
    }
    for (int idx = tid; idx < 16 * NPAD; idx += 256) c_lds[idx] = 0.0f;
    for (int idx = tid; idx < 14 * DIM; idx += 256) {
        int r = 2 + (idx >> 9), d = idx & (DIM - 1);
        o_lds[r * DIM + d] = 0.0f;
        ob_lds[r * OBPITCH + d] = (__bf16)0.0f;
    }
    for (int idx = tid; idx < 2 * DIM; idx += 256) {
        int k = idx >> 9, d = idx & (DIM - 1);
        int src = (k == 0) ? 50 : 98;                 // init_idx = [50, 196//2]
        float v = ug[src * DIM + d];
        o_lds[k * DIM + d] = v;
        ob_lds[k * OBPITCH + d] = (__bf16)(-2.0f * v);
    }
    for (int idx = tid + N_PTS; idx < 208; idx += 256) ny_lds[idx] = 0.0f;
    if (tid < 2) { nx_lds[tid] = 0.0f; cs_lds[tid] = 0.0f; }
    __syncthreads();

    // ny[n] = sum_d u[n][d]^2 (from the same rounded bf16 the GEMM uses);
    // nx[k] = sum_d o[k][d]^2 in f32
    if (tid < N_PTS) {
        float s = 0.0f;
        for (int d = 0; d < DIM; ++d) { float v = (float)u_lds[tid * DPITCH + d]; s += v * v; }
        ny_lds[tid] = s;
    } else if (tid < N_PTS + 2) {
        int k = tid - N_PTS;
        float s = 0.0f;
        for (int d = 0; d < DIM; ++d) { float v = o_lds[k * DIM + d]; s += v * v; }
        nx_lds[k] = s;
    }
    __syncthreads();

    // ---------------- iterations ----------------
    for (int it = 0; it < N_ITERS; ++it) {
        // ---- phase 1: qq = nx + ny + (-2 o)·u^T via bf16 WMMA; softmax -> c ----
        {
            const int  t1   = wave + 8;
            const bool has1 = (t1 < 13);
            const int  n0a  = wave * 16;
            const int  n0b  = has1 ? t1 * 16 : 0;
            v8f acc0 = {};
            v8f acc1 = {};
            for (int k0 = 0; k0 < DIM; k0 += 32) {
                // A: 16x32 bf16 tile of (-2*o), pre-converted in ob_lds.
                // per-lane element i holds K = k0 + half*16 + i:
                // two contiguous 16B loads from row lrow.
                v8bf alo = *(const v8bf*)(ob_lds + lrow * OBPITCH + k0 + half * 16);
                v8bf ahi = *(const v8bf*)(ob_lds + lrow * OBPITCH + k0 + half * 16 + 8);
                v16bf a = __builtin_shufflevector(alo, ahi,
                            0, 1, 2, 3, 4, 5, 6, 7, 8, 9, 10, 11, 12, 13, 14, 15);
                // B: 32x16 bf16 tile of u^T: lane holds 16 consecutive K=d values
                // from row (n0+lrow) -> two contiguous 16B LDS loads.
                v16bf b0 = *(const v16bf*)(u_lds + (n0a + lrow) * DPITCH + k0 + half * 16);
                v16bf b1 = *(const v16bf*)(u_lds + (n0b + lrow) * DPITCH + k0 + half * 16);
                acc0 = __builtin_amdgcn_wmma_f32_16x16x32_bf16(false, a, false, b0,
                                                               (short)0, acc0, false, false);
                acc1 = __builtin_amdgcn_wmma_f32_16x16x32_bf16(false, a, false, b1,
                                                               (short)0, acc1, false, false);
            }
            const float nx0 = nx_lds[0], nx1 = nx_lds[1];
            #pragma unroll
            for (int which = 0; which < 2; ++which) {
                v8f  acc = which ? acc1 : acc0;
                int  n0  = which ? n0b  : n0a;
                bool act = which ? has1 : true;
                int n = n0 + lrow;
                float nyv = ny_lds[n];
                // rows 0 and 1 of the D tile live in acc[0], acc[1] for lanes 0-15
                float q0 = acc[0] + nx0 + nyv;
                float q1 = acc[1] + nx1 + nyv;
                float b0v = sqrtf(fmaxf(q0, EPS_F));
                float b1v = sqrtf(fmaxf(q1, EPS_F));
                float t0v = BETA_F * b0v, t1v = BETA_F * b1v;
                float mx = fmaxf(t0v, t1v);
                float e0 = __expf(t0v - mx), e1 = __expf(t1v - mx);
                float inv = 1.0f / (e0 + e1);
                float c0 = e0 * inv, c1 = e1 * inv;
                if (act && half == 0 && n < N_PTS) {
                    c_lds[0 * NPAD + n] = c0;
                    c_lds[1 * NPAD + n] = c1;
                    atomicAdd(&cs_lds[0], c0);
                    atomicAdd(&cs_lds[1], c1);
                }
            }
        }
        __syncthreads();
        if (tid < 2) nx_lds[tid] = 0.0f;
        __syncthreads();

        // ---- phase 2: o = (c · u) / csum via f32 WMMA 16x16x4 (K = n) ----
        {
            v8f acc[4] = {};
            for (int nk = 0; nk < NPAD; nk += 4) {
                int ka = nk + (half ? 2 : 0);
                // A: 16x4 f32 tile of c: v0 -> K=ka, v1 -> K=ka+1, row M = lrow
                v2f a = *(const v2f*)(c_lds + lrow * NPAD + ka);
                #pragma unroll
                for (int j = 0; j < 4; ++j) {
                    int d0 = (wave * 4 + j) * 16;
                    // B: 4x16 f32 tile of u: each lane reads one element of a
                    // contiguous u row (bf16 -> f32 widen), no transpose needed
                    v2f bb;
                    bb.x = (float)u_lds[(ka    ) * DPITCH + d0 + lrow];
                    bb.y = (float)u_lds[(ka + 1) * DPITCH + d0 + lrow];
                    acc[j] = __builtin_amdgcn_wmma_f32_16x16x4_f32(false, a, false, bb,
                                                                   (short)0, acc[j], false, false);
                }
            }
            float s0 = cs_lds[0], s1 = cs_lds[1];
            float i0 = (s0 > 0.0f) ? 1.0f / s0 : 0.0f;
            float i1 = (s1 > 0.0f) ? 1.0f / s1 : 0.0f;
            if (half == 0) {
                #pragma unroll
                for (int j = 0; j < 4; ++j) {
                    int d0 = (wave * 4 + j) * 16;
                    float v0 = acc[j][0] * i0;
                    float v1 = acc[j][1] * i1;
                    o_lds[0 * DIM + d0 + lrow] = v0;
                    o_lds[1 * DIM + d0 + lrow] = v1;
                    ob_lds[0 * OBPITCH + d0 + lrow] = (__bf16)(-2.0f * v0);
                    ob_lds[1 * OBPITCH + d0 + lrow] = (__bf16)(-2.0f * v1);
                    atomicAdd(&nx_lds[0], v0 * v0);
                    atomicAdd(&nx_lds[1], v1 * v1);
                }
            }
        }
        __syncthreads();
        if (tid < 2) cs_lds[tid] = 0.0f;
        __syncthreads();
    }

    // ---------------- output: o (2048,2,512) then c (2048,2,196) ----------------
    for (int idx = tid; idx < 2 * DIM; idx += 256) {
        int k = idx >> 9, d = idx & (DIM - 1);
        out[(size_t)b * (2 * DIM) + idx] = o_lds[k * DIM + d];
    }
    const size_t c_base = (size_t)batches * (2 * DIM);
    for (int idx = tid; idx < 2 * N_PTS; idx += 256) {
        int k = idx / N_PTS, n = idx - k * N_PTS;
        out[c_base + (size_t)b * (2 * N_PTS) + idx] = c_lds[k * NPAD + n];
    }
}

extern "C" void kernel_launch(void* const* d_in, const int* in_sizes, int n_in,
                              void* d_out, int out_size, void* d_ws, size_t ws_size,
                              hipStream_t stream) {
    const float* u  = (const float*)d_in[0];
    float* out      = (float*)d_out;
    int batches     = in_sizes[0] / (N_PTS * DIM);   // 2048
    (void)d_ws; (void)ws_size; (void)n_in; (void)out_size;
    hipFuncSetAttribute((const void*)cluster_kernel,
                        hipFuncAttributeMaxDynamicSharedMemorySize, LDS_BYTES);
    cluster_kernel<<<batches, 256, LDS_BYTES, stream>>>(u, out, batches);
}
